// HAN_82205674045527
// MI455X (gfx1250) — compile-verified, matrix operand order checked by reading, not run on hardware
//
#include <hip/hip_runtime.h>
#include <math.h>

// ---------- problem constants (match reference) ----------
#define NN   100000   // nodes
#define FIN  1024     // input features
#define HH   8        // heads
#define CC   128      // hidden channels
#define EE   600000   // edges per metapath
#define PP   3        // metapaths
#define OUTC 3        // classes

typedef __attribute__((ext_vector_type(16))) __bf16 v16bf;
typedef __attribute__((ext_vector_type(8)))  float  v8f;

// gfx1250 async direct-to-LDS path (guarded: falls back to VGPR round-trip copy)
#if defined(__has_builtin)
#if __has_builtin(__builtin_amdgcn_global_load_async_to_lds_b128) && \
    __has_builtin(__builtin_amdgcn_s_wait_asynccnt)
#define HAN_ASYNC_LDS 1
// builtin expects int4 pointers: global (AS1) source, LDS (AS3) destination
typedef int han_i4 __attribute__((vector_size(16)));
typedef __attribute__((address_space(1))) han_i4* han_gp_t;
typedef __attribute__((address_space(3))) han_i4* han_lp_t;
#endif
#endif

// ---------- workspace layout (bytes) ----------
static constexpr size_t OFF_H    = 0;                                   // [N,C] f32
static constexpr size_t OFF_OUTS = OFF_H    + (size_t)NN*CC*4;          // [P,N,C] f32
static constexpr size_t OFF_ALS  = OFF_OUTS + (size_t)PP*NN*CC*4;       // [P,N,H] f32
static constexpr size_t OFF_ALD  = OFF_ALS  + (size_t)PP*NN*HH*4;       // [P,N,H] f32
static constexpr size_t OFF_MMAX = OFF_ALD  + (size_t)PP*NN*HH*4;       // [P,N,H] u32 (keyed max)
static constexpr size_t OFF_DEN  = OFF_MMAX + (size_t)PP*NN*HH*4;       // [P,N,H] f32
static constexpr size_t OFF_EBUF = OFF_DEN  + (size_t)PP*NN*HH*4;       // [P,E,H] f32
static constexpr size_t OFF_WP   = OFF_EBUF + (size_t)PP*EE*HH*4;       // packed W bf16 [1024*128]
static constexpr size_t OFF_KWP  = OFF_WP   + (size_t)FIN*CC*2;         // packed kW bf16 [128*128]
static constexpr size_t OFF_WSEM = OFF_KWP  + (size_t)CC*CC*2;          // [P,C] f32
static constexpr size_t OFF_BETA = OFF_WSEM + (size_t)PP*CC*4;          // [P] f32

// order-preserving float -> uint key for atomicMax
__device__ __forceinline__ unsigned fkey(float f) {
  unsigned u = __float_as_uint(f);
  return (u & 0x80000000u) ? ~u : (u | 0x80000000u);
}
__device__ __forceinline__ float funkey(unsigned u) {
  return (u & 0x80000000u) ? __uint_as_float(u & 0x7fffffffu) : __uint_as_float(~u);
}

// ---------- utility init kernels ----------
__global__ void han_zero4(float4* __restrict__ p, int n4) {
  int i = blockIdx.x * 256 + threadIdx.x;
  if (i < n4) { float4 z = {0.f, 0.f, 0.f, 0.f}; p[i] = z; }
}
__global__ void han_fillu(unsigned* __restrict__ p, int n, unsigned v) {
  int i = blockIdx.x * 256 + threadIdx.x;
  if (i < n) p[i] = v;
}

// ---------- pack f32 [K,128] row-major into CDNA5 bf16 B-fragment layout ----------
// dst[((kb*8+nt)*32+lane)*16+j]; lane<16 : K in {0..7,16..23}; lane>=16 : {8..15,24..31}
__global__ void han_pack_bf16(const float* __restrict__ src, __bf16* __restrict__ dst, int K) {
  int id = blockIdx.x * 256 + threadIdx.x;
  int total = K * CC;
  if (id >= total) return;
  int j    = id & 15;
  int lane = (id >> 4) & 31;
  int nt   = (id >> 9) & 7;
  int kb   = id >> 12;
  int k = kb * 32 + ((lane >> 4) << 3) + (j < 8 ? j : j + 8);
  int n = nt * 16 + (lane & 15);
  dst[id] = (__bf16)src[k * CC + n];
}

// ---------- projection GEMM: h = x @ W + b  via v_wmma_f32_16x16x32_bf16 ----------
// 4 waves/block, each wave -> 16 rows x 128 cols; W tile staged in LDS per k-block
// (async direct-to-LDS when available). A(kb+1) global loads issued before the
// 8 WMMAs of kb; B fragments register double-buffered so ds_load(nt+1) overlaps wmma(nt).
__global__ void __launch_bounds__(128)
han_proj_wmma(const float* __restrict__ x, const __bf16* __restrict__ Wp,
              const float* __restrict__ bproj, float* __restrict__ hout) {
  __shared__ __align__(16) __bf16 ldsW[8 * 32 * 16];   // 8 KB
  int lane    = threadIdx.x & 31;
  int wv      = threadIdx.x >> 5;
  int rowbase = blockIdx.x * 64 + wv * 16;
  int m       = rowbase + (lane & 15);
  int mc      = (m < NN) ? m : (NN - 1);               // clamp OOB rows (stores are guarded)
  int khalf   = (lane >> 4) << 3;
  const float* xrow = x + (size_t)mc * FIN;

  v8f acc[8];
  v8f z = {0.f, 0.f, 0.f, 0.f, 0.f, 0.f, 0.f, 0.f};
#pragma unroll
  for (int t = 0; t < 8; ++t) acc[t] = z;

  // prologue: A fragment for kb=0
  float4 f0, f1, f2, f3;
  {
    const float4* xr = reinterpret_cast<const float4*>(xrow + khalf);
    f0 = xr[0]; f1 = xr[1]; f2 = xr[4]; f3 = xr[5];
  }

  for (int kb = 0; kb < FIN / 32; ++kb) {
    __syncthreads();
#ifdef HAN_ASYNC_LDS
    {   // async memory->LDS fill: 4 x b128 per thread, no VGPR round-trip
      const char* gsrc = reinterpret_cast<const char*>(Wp) + (size_t)kb * 8192;
      unsigned lbase = (unsigned)(size_t)&ldsW[0];
#pragma unroll
      for (int i = 0; i < 4; ++i) {
        unsigned off = (threadIdx.x + i * 128) * 16;
        han_gp_t gp = (han_gp_t)(unsigned long long)(size_t)(gsrc + off);
        han_lp_t lp = (han_lp_t)(size_t)(lbase + off);
        __builtin_amdgcn_global_load_async_to_lds_b128(gp, lp, 0, 0);
      }
      __builtin_amdgcn_s_wait_asynccnt(0);
    }
#else
    {   // fallback: cooperative copy through VGPRs
      const uint4* gw = reinterpret_cast<const uint4*>(Wp) + (size_t)kb * 512;
      uint4* lw = reinterpret_cast<uint4*>(ldsW);
#pragma unroll
      for (int i = 0; i < 4; ++i) lw[threadIdx.x + i * 128] = gw[threadIdx.x + i * 128];
    }
#endif
    __syncthreads();

    // convert current A fragment
    v16bf a;
    a[0]  = (__bf16)f0.x; a[1]  = (__bf16)f0.y; a[2]  = (__bf16)f0.z; a[3]  = (__bf16)f0.w;
    a[4]  = (__bf16)f1.x; a[5]  = (__bf16)f1.y; a[6]  = (__bf16)f1.z; a[7]  = (__bf16)f1.w;
    a[8]  = (__bf16)f2.x; a[9]  = (__bf16)f2.y; a[10] = (__bf16)f2.z; a[11] = (__bf16)f2.w;
    a[12] = (__bf16)f3.x; a[13] = (__bf16)f3.y; a[14] = (__bf16)f3.z; a[15] = (__bf16)f3.w;

    // prefetch next A fragment (waited on at next iteration's convert)
    if (kb + 1 < FIN / 32) {
      const float4* xr = reinterpret_cast<const float4*>(xrow + (kb + 1) * 32 + khalf);
      f0 = xr[0]; f1 = xr[1]; f2 = xr[4]; f3 = xr[5];
    }

    // register double-buffered B: load(nt+1) issued before wmma(nt)
    const __bf16* lb = &ldsW[lane * 16];
    v16bf bcur = *reinterpret_cast<const v16bf*>(lb);
#pragma unroll
    for (int nt = 0; nt < 8; ++nt) {
      v16bf bnext = bcur;
      if (nt < 7) bnext = *reinterpret_cast<const v16bf*>(lb + (nt + 1) * 512);
      acc[nt] = __builtin_amdgcn_wmma_f32_16x16x32_bf16(false, a, false, bcur,
                                                        (short)0, acc[nt], false, false);
      bcur = bnext;
    }
  }

#pragma unroll
  for (int nt = 0; nt < 8; ++nt) {
    int col = nt * 16 + (lane & 15);
    float bias = bproj[col];
#pragma unroll
    for (int r = 0; r < 8; ++r) {
      int mr = rowbase + r + khalf;        // lanes>=16 hold rows M=r+8
      if (mr < NN) hout[(size_t)mr * CC + col] = acc[nt][r] + bias;
    }
  }
}

// ---------- per-node attention logits: al_s/al_d = sum_d h*a ----------
__global__ void han_alpha(const float* __restrict__ h, const float* __restrict__ asrc,
                          const float* __restrict__ adst, float* __restrict__ als,
                          float* __restrict__ ald) {
  int id = blockIdx.x * 256 + threadIdx.x;
  if (id >= PP * NN) return;
  int p = id / NN, n = id - p * NN;
  const float* hr = h + (size_t)n * CC;
#pragma unroll
  for (int hh = 0; hh < HH; ++hh) {
    float s = 0.f, d = 0.f;
#pragma unroll
    for (int j = 0; j < 16; ++j) {
      float v = hr[hh * 16 + j];
      s += v * asrc[(p * HH + hh) * 16 + j];
      d += v * adst[(p * HH + hh) * 16 + j];
    }
    als[(size_t)id * HH + hh] = s;
    ald[(size_t)id * HH + hh] = d;
  }
}

// ---------- edge pass 1: e = leaky_relu(al_s[src]+al_d[dst]); segment max ----------
__global__ void han_edge_max(const long long* __restrict__ ei, const float* __restrict__ als,
                             const float* __restrict__ ald, float* __restrict__ ebuf,
                             unsigned* __restrict__ mmax) {
  int id = blockIdx.x * 256 + threadIdx.x;
  if (id >= PP * EE) return;
  int p = id / EE, e = id - p * EE;
  int src = (int)ei[(size_t)(p * 2 + 0) * EE + e];
  int dst = (int)ei[(size_t)(p * 2 + 1) * EE + e];
  const float* as = als + ((size_t)p * NN + src) * HH;
  const float* ad = ald + ((size_t)p * NN + dst) * HH;
  unsigned* mm = mmax + ((size_t)p * NN + dst) * HH;
  float* eb = ebuf + (size_t)id * HH;
#pragma unroll
  for (int hh = 0; hh < HH; ++hh) {
    float v = as[hh] + ad[hh];
    v = (v > 0.f) ? v : 0.2f * v;
    eb[hh] = v;
    atomicMax(&mm[hh], fkey(v));
  }
}

// ---------- edge pass 2: p = exp(e - m[dst]); segment sum ----------
__global__ void han_edge_sum(const long long* __restrict__ ei, const unsigned* __restrict__ mmax,
                             float* __restrict__ ebuf, float* __restrict__ denom) {
  int id = blockIdx.x * 256 + threadIdx.x;
  if (id >= PP * EE) return;
  int p = id / EE, e = id - p * EE;
  int dst = (int)ei[(size_t)(p * 2 + 1) * EE + e];
  const unsigned* mm = mmax + ((size_t)p * NN + dst) * HH;
  float* dn = denom + ((size_t)p * NN + dst) * HH;
  float* eb = ebuf + (size_t)id * HH;
#pragma unroll
  for (int hh = 0; hh < HH; ++hh) {
    float pv = __expf(eb[hh] - funkey(mm[hh]));
    eb[hh] = pv;
    atomicAdd(&dn[hh], pv);
  }
}

// ---------- edge pass 3: out[dst] += h[src] * alpha ----------
__global__ void han_edge_scatter(const long long* __restrict__ ei, const float* __restrict__ h,
                                 const float* __restrict__ ebuf, const float* __restrict__ denom,
                                 float* __restrict__ outs) {
  int id = blockIdx.x * 256 + threadIdx.x;       // over P*E*H
  if (id >= PP * EE * HH) return;
  int hh = id & 7;
  int pe = id >> 3;
  int p = pe / EE, e = pe - p * EE;
  int src = (int)ei[(size_t)(p * 2 + 0) * EE + e];
  int dst = (int)ei[(size_t)(p * 2 + 1) * EE + e];
  float alpha = ebuf[(size_t)pe * HH + hh] /
                (denom[((size_t)p * NN + dst) * HH + hh] + 1e-16f);
  const float4* hs = reinterpret_cast<const float4*>(h + (size_t)src * CC + hh * 16);
  float* od = outs + ((size_t)p * NN + dst) * CC + hh * 16;
#pragma unroll
  for (int q4 = 0; q4 < 4; ++q4) {
    float4 v = hs[q4];
    atomicAdd(&od[q4 * 4 + 0], v.x * alpha);
    atomicAdd(&od[q4 * 4 + 1], v.y * alpha);
    atomicAdd(&od[q4 * 4 + 2], v.z * alpha);
    atomicAdd(&od[q4 * 4 + 3], v.w * alpha);
  }
}

// ---------- semantic attention: wsem[p,:] += sum_n tanh(relu(outs) @ kW + kb) ----------
// 4 waves/block, one 16-node tile per wave; relu fused into A build;
// B fragments preloaded per k-block; tanh + row-reduce in epilogue.
__global__ void __launch_bounds__(128)
han_semantic_wmma(const float* __restrict__ outs, const __bf16* __restrict__ kWp,
                  const float* __restrict__ kbias, float* __restrict__ wsem) {
  int lane = threadIdx.x & 31;
  int wv   = threadIdx.x >> 5;
  int p    = blockIdx.y;
  int tile = blockIdx.x * 4 + wv;
  if (tile >= NN / 16) return;                   // wave-uniform guard
  int m = tile * 16 + (lane & 15);               // N divisible by 16
  int khalf = (lane >> 4) << 3;
  const float* arow = outs + ((size_t)p * NN + m) * CC;

  v8f acc[8];
  v8f z = {0.f, 0.f, 0.f, 0.f, 0.f, 0.f, 0.f, 0.f};
#pragma unroll
  for (int t = 0; t < 8; ++t) acc[t] = z;

  for (int kb = 0; kb < CC / 32; ++kb) {
    // preload all 8 B fragments from global (kW packed = 32 KB, L2/L0 resident)
    v16bf bfr[8];
#pragma unroll
    for (int nt = 0; nt < 8; ++nt)
      bfr[nt] = *reinterpret_cast<const v16bf*>(kWp + ((size_t)(kb * 8 + nt) * 32 + lane) * 16);

    int k0 = kb * 32 + khalf;
    v16bf a;
#pragma unroll
    for (int j = 0; j < 8; ++j) {
      float v0 = arow[k0 + j];       v0 = (v0 > 0.f) ? v0 : 0.f;   // relu at read
      float v1 = arow[k0 + 16 + j];  v1 = (v1 > 0.f) ? v1 : 0.f;
      a[j] = (__bf16)v0;  a[8 + j] = (__bf16)v1;
    }
#pragma unroll
    for (int nt = 0; nt < 8; ++nt)
      acc[nt] = __builtin_amdgcn_wmma_f32_16x16x32_bf16(false, a, false, bfr[nt],
                                                        (short)0, acc[nt], false, false);
  }
#pragma unroll
  for (int nt = 0; nt < 8; ++nt) {
    int col = nt * 16 + (lane & 15);
    float bias = kbias[col];
    float s = 0.f;
#pragma unroll
    for (int r = 0; r < 8; ++r) s += tanhf(acc[nt][r] + bias);
    atomicAdd(&wsem[p * CC + col], s);
  }
}

// ---------- semantic softmax over P ----------
__global__ void han_beta(const float* __restrict__ wsem, const float* __restrict__ q,
                         float* __restrict__ beta) {
  if (threadIdx.x != 0 || blockIdx.x != 0) return;
  float sc[PP];
  float mx = -1e30f;
  for (int p = 0; p < PP; ++p) {
    float s = 0.f;
    for (int c = 0; c < CC; ++c) s += q[c] * wsem[p * CC + c];
    sc[p] = s * (1.0f / (float)NN);
    mx = fmaxf(mx, sc[p]);
  }
  float den = 0.f;
  for (int p = 0; p < PP; ++p) { sc[p] = __expf(sc[p] - mx); den += sc[p]; }
  for (int p = 0; p < PP; ++p) beta[p] = sc[p] / den;
}

// ---------- fuse + classifier: out = (sum_p beta_p * relu(outs_p)) @ linW + linb ----------
__global__ void han_final(const float* __restrict__ outs, const float* __restrict__ beta,
                          const float* __restrict__ linW, const float* __restrict__ linb,
                          float* __restrict__ out) {
  int n = blockIdx.x * 256 + threadIdx.x;
  if (n >= NN) return;
  float b0 = beta[0], b1 = beta[1], b2 = beta[2];
  const float* o0 = outs + (size_t)n * CC;
  const float* o1 = outs + ((size_t)NN + n) * CC;
  const float* o2 = outs + ((size_t)2 * NN + n) * CC;
  float a0 = linb[0], a1 = linb[1], a2 = linb[2];
#pragma unroll 4
  for (int c = 0; c < CC; ++c) {
    float f = b0 * fmaxf(o0[c], 0.f) + b1 * fmaxf(o1[c], 0.f) + b2 * fmaxf(o2[c], 0.f);
    a0 += f * linW[c * OUTC + 0];
    a1 += f * linW[c * OUTC + 1];
    a2 += f * linW[c * OUTC + 2];
  }
  out[n * OUTC + 0] = a0;
  out[n * OUTC + 1] = a1;
  out[n * OUTC + 2] = a2;
}

extern "C" void kernel_launch(void* const* d_in, const int* in_sizes, int n_in,
                              void* d_out, int out_size, void* d_ws, size_t ws_size,
                              hipStream_t stream) {
  (void)in_sizes; (void)n_in; (void)out_size; (void)ws_size;
  const float*     x     = (const float*)d_in[0];
  const long long* ei    = (const long long*)d_in[1];   // int64 edge index
  const float*     Wproj = (const float*)d_in[2];
  const float*     bproj = (const float*)d_in[3];
  const float*     asrc  = (const float*)d_in[4];
  const float*     adst  = (const float*)d_in[5];
  const float*     kW    = (const float*)d_in[6];
  const float*     kb    = (const float*)d_in[7];
  const float*     q     = (const float*)d_in[8];
  const float*     linW  = (const float*)d_in[9];
  const float*     linb  = (const float*)d_in[10];
  float* out = (float*)d_out;

  char* ws = (char*)d_ws;
  float*    h     = (float*)(ws + OFF_H);
  float*    outs  = (float*)(ws + OFF_OUTS);
  float*    als   = (float*)(ws + OFF_ALS);
  float*    ald   = (float*)(ws + OFF_ALD);
  unsigned* mmax  = (unsigned*)(ws + OFF_MMAX);
  float*    denom = (float*)(ws + OFF_DEN);
  float*    ebuf  = (float*)(ws + OFF_EBUF);
  __bf16*   Wp    = (__bf16*)(ws + OFF_WP);
  __bf16*   kWp   = (__bf16*)(ws + OFF_KWP);
  float*    wsem  = (float*)(ws + OFF_WSEM);
  float*    beta  = (float*)(ws + OFF_BETA);

  // pack weights to bf16 fragment layout
  han_pack_bf16<<<(FIN * CC + 255) / 256, 256, 0, stream>>>(Wproj, Wp, FIN);
  han_pack_bf16<<<(CC * CC + 255) / 256, 256, 0, stream>>>(kW, kWp, CC);

  // init accumulators
  han_zero4<<<((PP * NN * CC / 4) + 255) / 256, 256, 0, stream>>>((float4*)outs, PP * NN * CC / 4);
  han_zero4<<<((PP * NN * HH / 4) + 255) / 256, 256, 0, stream>>>((float4*)denom, PP * NN * HH / 4);
  han_zero4<<<((PP * CC / 4) + 255) / 256, 256, 0, stream>>>((float4*)wsem, PP * CC / 4);
  han_fillu<<<((PP * NN * HH) + 255) / 256, 256, 0, stream>>>(mmax, PP * NN * HH, 0x007FFFFFu); // key(-inf)

  // 1) projection GEMM (WMMA bf16, async LDS staging)
  han_proj_wmma<<<(NN + 63) / 64, 128, 0, stream>>>(x, Wp, bproj, h);

  // 2) GAT attention per metapath
  han_alpha<<<(PP * NN + 255) / 256, 256, 0, stream>>>(h, asrc, adst, als, ald);
  han_edge_max<<<(PP * EE + 255) / 256, 256, 0, stream>>>(ei, als, ald, ebuf, mmax);
  han_edge_sum<<<(PP * EE + 255) / 256, 256, 0, stream>>>(ei, mmax, ebuf, denom);
  han_edge_scatter<<<(PP * EE * HH + 255) / 256, 256, 0, stream>>>(ei, h, ebuf, denom, outs);

  // 3) semantic attention (WMMA bf16) + softmax + fuse/classify
  han_semantic_wmma<<<dim3((NN / 16 + 3) / 4, PP), 128, 0, stream>>>(outs, kWp, kb, wsem);
  han_beta<<<1, 1, 0, stream>>>(wsem, q, beta);
  han_final<<<(NN + 255) / 256, 256, 0, stream>>>(outs, beta, linW, linb, out);
}
